// NeuroBiMambaBlock_69758858821818
// MI455X (gfx1250) — matrix operand order repaired and checked by reading, not compiled, not run on hardware
//
#include <hip/hip_runtime.h>
#include <hip/hip_bf16.h>
#include <math.h>

// ---------------------------------------------------------------------------
// NeuroBiMambaBlock for MI455X (gfx1250, wave32, WMMA).
// GEMMs: v_wmma_f32_16x16x32_bf16; B tiles staged in LDS via
// global_load_async_to_lds_b128 (ASYNCcnt double buffer), A tiles
// register-double-buffered so no load gates its consuming WMMA.
// Sequence direction handled by `rev` flag on dwconv + scan only.
// ---------------------------------------------------------------------------

#define BATCH 4
#define SEQL  2048
#define DMODEL 512
#define DINNER 1024
#define MROWS (BATCH * SEQL)   // 8192

typedef __attribute__((ext_vector_type(16))) __bf16 v16bf;
typedef __attribute__((ext_vector_type(8)))  float  v8f;

__device__ __forceinline__ unsigned short f32_to_bf16(float f) {
    unsigned int u = __float_as_uint(f);
    unsigned int r = u + 0x7fffu + ((u >> 16) & 1u);   // round-to-nearest-even
    return (unsigned short)(r >> 16);
}
__device__ __forceinline__ float siluf(float x) {
    return x / (1.f + __expf(-x));
}
__device__ __forceinline__ float softplusf(float x) {
    return (x > 20.f) ? x : log1pf(__expf(x));
}

// ---------------------------------------------------------------------------
// f32 -> bf16 conversion
// ---------------------------------------------------------------------------
__global__ void cvt_bf16_kernel(const float* __restrict__ src,
                                unsigned short* __restrict__ dst, int n) {
    int i = blockIdx.x * 256 + threadIdx.x;
    if (i < n) dst[i] = f32_to_bf16(src[i]);
}

// ---------------------------------------------------------------------------
// LayerNorm over DM=512, output bf16. One block (256 thr) per row.
// ---------------------------------------------------------------------------
__global__ void layernorm_bf16_kernel(const float* __restrict__ x,
                                      const float* __restrict__ g,
                                      const float* __restrict__ b,
                                      unsigned short* __restrict__ out) {
    const int row = blockIdx.x;
    const float* xr = x + (size_t)row * DMODEL;
    float v0 = xr[threadIdx.x], v1 = xr[threadIdx.x + 256];
    __shared__ float rs[256], rs2[256];
    rs[threadIdx.x]  = v0 + v1;
    rs2[threadIdx.x] = v0 * v0 + v1 * v1;
    __syncthreads();
    for (int off = 128; off > 0; off >>= 1) {
        if (threadIdx.x < off) {
            rs[threadIdx.x]  += rs[threadIdx.x + off];
            rs2[threadIdx.x] += rs2[threadIdx.x + off];
        }
        __syncthreads();
    }
    float mu   = rs[0] * (1.f / DMODEL);
    float var  = rs2[0] * (1.f / DMODEL) - mu * mu;
    float rstd = rsqrtf(var + 1e-5f);
    int i0 = threadIdx.x, i1 = threadIdx.x + 256;
    out[(size_t)row * DMODEL + i0] = f32_to_bf16((v0 - mu) * rstd * g[i0] + b[i0]);
    out[(size_t)row * DMODEL + i1] = f32_to_bf16((v1 - mu) * rstd * g[i1] + b[i1]);
}

// ---------------------------------------------------------------------------
// Shared epilogue for GEMM kernels.
//   EP=0 : Cf[m*N+n] = acc
//   EP=1 : Cb[m*outLd + outColOff + n] = bf16(acc * extra[m*N+n])   (gate g)
//   EP=2 : Cf[m*N+n] = acc + extra[m*N+n]                           (residual)
//   EP=3 : Cf[m*N+n] = softplus(acc + extra[n])                     (dt path)
// ---------------------------------------------------------------------------
template<int EP>
__device__ __forceinline__ void gemm_epilogue(float v, int m, int n, int N,
                                              float* __restrict__ Cf,
                                              unsigned short* __restrict__ Cb,
                                              const float* __restrict__ extra,
                                              int outLd, int outColOff) {
    if (EP == 0) {
        Cf[(size_t)m * N + n] = v;
    } else if (EP == 1) {
        v *= extra[(size_t)m * N + n];
        Cb[(size_t)m * outLd + outColOff + n] = f32_to_bf16(v);
    } else if (EP == 2) {
        Cf[(size_t)m * N + n] = v + extra[(size_t)m * N + n];
    } else { // EP == 3
        Cf[(size_t)m * N + n] = softplusf(v + extra[n]);
    }
}

// A-tile register image: 16x32 bf16 per wave (8 dwords / lane).
struct ATile { unsigned int u[8]; };

__device__ __forceinline__ void load_a_tile(const unsigned short* __restrict__ Ar,
                                            int half, ATile& a) {
    // lanes 0-15: K0-7 (u0-3) & K16-23 (u4-7); lanes 16-31: K8-15 & K24-31
#pragma unroll
    for (int r = 0; r < 4; ++r)
        a.u[r] = *(const unsigned int*)(Ar + 8 * half + 2 * r);
#pragma unroll
    for (int r = 0; r < 4; ++r)
        a.u[4 + r] = *(const unsigned int*)(Ar + 16 + 8 * half + 2 * r);
}

// ---------------------------------------------------------------------------
// Main WMMA GEMM, fully pipelined:
//   - B tiles (64 n x 32 k = 4KB) DMA'd to LDS with
//     global_load_async_to_lds_b128 (16B/thread), double buffered,
//     synchronized via s_wait_asynccnt + barriers.
//   - A tiles register-double-buffered: the A load for step i+1 issues
//     before the ds_loads/WMMAs of step i, hiding global latency.
// Block = 256 thr = 8 waves; block tile 128(M) x 64(N); wave = 16x64.
// Requires M % 128 == 0, N % 64 == 0, K % 64 == 0 (nsteps even).
// ---------------------------------------------------------------------------
template<int EP>
__global__ void gemm_wmma_lds_kernel(const unsigned short* __restrict__ A,
                                     const unsigned short* __restrict__ W,
                                     float* __restrict__ Cf,
                                     unsigned short* __restrict__ Cb,
                                     const float* __restrict__ extra,
                                     int M, int N, int K,
                                     int outLd, int outColOff) {
    constexpr int PITCH = 80;                 // B row pitch in LDS (bank-skewed, 16B mult)
    constexpr int BUFB  = 64 * PITCH;         // 5120 B per buffer
    __shared__ __align__(16) unsigned char sB[2 * BUFB];

    const int tid  = threadIdx.x;
    const int lane = tid & 31;
    const int wave = tid >> 5;                // 0..7 -> M strip
    const int half = lane >> 4;
    const int l15  = lane & 15;
    const int mBase = blockIdx.y * 128 + wave * 16;
    const int nBase = blockIdx.x * 64;
    const int mRow  = mBase + l15;

    // async-fill mapping: 256 threads x 16B cover the 64x32 bf16 B tile
    const int fn = tid >> 2;                  // B row in tile (0..63)
    const int fs = tid & 3;                   // 16B chunk within the 64B row slice
    const unsigned int ldsBase =
        (unsigned int)(uintptr_t)(&sB[0]) + (unsigned int)(fn * PITCH + fs * 16);
    const unsigned int dstBuf0 = ldsBase;
    const unsigned int dstBuf1 = ldsBase + (unsigned int)BUFB;
    const unsigned short* gW = W + (size_t)(nBase + fn) * K + fs * 8;
    const unsigned short* gA = A + (size_t)mRow * K;

    const int nsteps = K >> 5;                // even by construction

    v8f acc[4];
#pragma unroll
    for (int t = 0; t < 4; ++t)
#pragma unroll
        for (int r = 0; r < 8; ++r) acc[t][r] = 0.f;

    // prologue: B tile 0 -> buf0 (async), A tile 0 -> registers
    {
        unsigned long long ga = (unsigned long long)(uintptr_t)gW;
        asm volatile("global_load_async_to_lds_b128 %0, %1, off"
                     :: "v"(dstBuf0), "v"(ga) : "memory");
    }
    ATile aR, aN;
    load_a_tile(gA, half, aR);

    auto compute = [&](const unsigned char* __restrict__ sbuf, const ATile& a) {
        v16bf av;
        {
            union { unsigned int u[8]; v16bf v; } cvt_;
#pragma unroll
            for (int r = 0; r < 8; ++r) cvt_.u[r] = a.u[r];
            av = cvt_.v;
        }
#pragma unroll
        for (int t = 0; t < 4; ++t) {
            union { uint4 q[2]; v16bf v; } bm;
            const unsigned char* bp = sbuf + (t * 16 + l15) * PITCH + 16 * half;
            bm.q[0] = *(const uint4*)(bp);
            bm.q[1] = *(const uint4*)(bp + 32);
            acc[t] = __builtin_amdgcn_wmma_f32_16x16x32_bf16(
                false, av, false, bm.v, (short)0, acc[t], false, false);
        }
    };

    for (int i = 0; i < nsteps; i += 2) {
        // ---- phase A: consume buf0 (step i), prefetch step i+1 ----
        {
            unsigned long long ga =
                (unsigned long long)(uintptr_t)(gW + (size_t)(i + 1) * 32);
            asm volatile("global_load_async_to_lds_b128 %0, %1, off"
                         :: "v"(dstBuf1), "v"(ga) : "memory");
            asm volatile("s_wait_asynccnt 0x1" ::: "memory");
        }
        __syncthreads();                       // buf0 (step i) fully written
        load_a_tile(gA + (size_t)(i + 1) * 32, half, aN);   // A for step i+1
        compute(&sB[0], aR);
        __syncthreads();                       // buf0 free for refill

        // ---- phase B: consume buf1 (step i+1), prefetch step i+2 ----
        const bool more = (i + 2 < nsteps);
        if (more) {
            unsigned long long ga =
                (unsigned long long)(uintptr_t)(gW + (size_t)(i + 2) * 32);
            asm volatile("global_load_async_to_lds_b128 %0, %1, off"
                         :: "v"(dstBuf0), "v"(ga) : "memory");
            asm volatile("s_wait_asynccnt 0x1" ::: "memory");
        } else {
            asm volatile("s_wait_asynccnt 0x0" ::: "memory");
        }
        __syncthreads();                       // buf1 (step i+1) fully written
        if (more)
            load_a_tile(gA + (size_t)(i + 2) * 32, half, aR);  // A for step i+2
        compute(&sB[BUFB], aN);
        __syncthreads();                       // buf1 free for refill
    }

    // C/D layout: lanes 0-15 -> n=lane, m=vgpr r; lanes 16-31 -> m=r+8
#pragma unroll
    for (int t = 0; t < 4; ++t)
#pragma unroll
        for (int r = 0; r < 8; ++r)
            gemm_epilogue<EP>(acc[t][r], mBase + r + 8 * half,
                              nBase + t * 16 + l15, N, Cf, Cb, extra,
                              outLd, outColOff);
}

// ---------------------------------------------------------------------------
// Simple direct-load WMMA GEMM (for the small N=96 xdbl projection).
// Block = 128 thr = 4 waves; wave = 16 rows x 16*NT cols.
// ---------------------------------------------------------------------------
template<int NT, int EP>
__global__ void gemm_wmma_kernel(const unsigned short* __restrict__ A,
                                 const unsigned short* __restrict__ W,
                                 float* __restrict__ Cf,
                                 unsigned short* __restrict__ Cb,
                                 const float* __restrict__ extra,
                                 int M, int N, int K,
                                 int outLd, int outColOff) {
    const int lane  = threadIdx.x & 31;
    const int wave  = threadIdx.x >> 5;
    const int half  = lane >> 4;
    const int l15   = lane & 15;
    const int mBase = blockIdx.y * 64 + wave * 16;
    const int nBase = blockIdx.x * (16 * NT);
    const int mRow  = mBase + l15;

    v8f acc[NT];
#pragma unroll
    for (int t = 0; t < NT; ++t)
#pragma unroll
        for (int r = 0; r < 8; ++r) acc[t][r] = 0.f;

    for (int k0 = 0; k0 < K; k0 += 32) {
        union { unsigned int u[8]; v16bf v; } a;
        const unsigned short* Ar = A + (size_t)mRow * K + k0;
#pragma unroll
        for (int r = 0; r < 4; ++r)
            a.u[r] = *(const unsigned int*)(Ar + 8 * half + 2 * r);
#pragma unroll
        for (int r = 0; r < 4; ++r)
            a.u[4 + r] = *(const unsigned int*)(Ar + 16 + 8 * half + 2 * r);

#pragma unroll
        for (int t = 0; t < NT; ++t) {
            union { unsigned int u[8]; v16bf v; } bm;
            const unsigned short* Wr =
                W + (size_t)(nBase + t * 16 + l15) * K + k0 + 8 * half;
#pragma unroll
            for (int r = 0; r < 8; ++r)
                bm.u[r] = *(const unsigned int*)(Wr + 2 * r + 8 * (r >> 2));
            acc[t] = __builtin_amdgcn_wmma_f32_16x16x32_bf16(
                false, a.v, false, bm.v, (short)0, acc[t], false, false);
        }
    }
#pragma unroll
    for (int t = 0; t < NT; ++t)
#pragma unroll
        for (int r = 0; r < 8; ++r)
            gemm_epilogue<EP>(acc[t][r], mBase + r + 8 * half,
                              nBase + t * 16 + l15, N, Cf, Cb, extra,
                              outLd, outColOff);
}

// ---------------------------------------------------------------------------
// g = silu(xp[:, 1024:2048])
// ---------------------------------------------------------------------------
__global__ void silu_half_kernel(const float* __restrict__ xp,
                                 float* __restrict__ g) {
    size_t i = (size_t)blockIdx.x * 256 + threadIdx.x;   // over M*1024
    size_t row = i >> 10;
    int c = (int)(i & 1023);
    g[i] = siluf(xp[row * 2048 + 1024 + c]);
}

// ---------------------------------------------------------------------------
// Causal (or time-reversed causal) depthwise conv K=4 + SiLU.
// fwd:  y[l] = b + sum_k w[k] * x[l-3+k]
// rev:  y[l] = b + sum_k w[k] * x[l+3-k]   (== flip . causal . flip)
// ---------------------------------------------------------------------------
__global__ void dwconv_silu_kernel(const float* __restrict__ in, int inLd,
                                   const float* __restrict__ w,
                                   const float* __restrict__ bias,
                                   float* __restrict__ outF,
                                   unsigned short* __restrict__ outB, int rev) {
    size_t idx = (size_t)blockIdx.x * 256 + threadIdx.x;   // over B*L*DINNER
    int c = (int)(idx & (DINNER - 1));
    size_t bl = idx >> 10;
    int l = (int)(bl & (SEQL - 1));
    int b = (int)(bl >> 11);
    float acc = bias[c];
#pragma unroll
    for (int k = 0; k < 4; ++k) {
        int li = rev ? (l + 3 - k) : (l - 3 + k);
        if (li >= 0 && li < SEQL)
            acc += w[c * 4 + k] * in[((size_t)b * SEQL + li) * inLd + c];
    }
    float y = siluf(acc);
    outF[idx] = y;
    outB[idx] = f32_to_bf16(y);
}

// ---------------------------------------------------------------------------
// dtr (first 64 cols of xdbl) -> bf16
// ---------------------------------------------------------------------------
__global__ void extract_dtr_kernel(const float* __restrict__ xdbl,
                                   unsigned short* __restrict__ d) {
    size_t i = (size_t)blockIdx.x * 256 + threadIdx.x;   // over M*64
    size_t row = i >> 6;
    int c = (int)(i & 63);
    d[i] = f32_to_bf16(xdbl[row * 96 + c]);
}

// ---------------------------------------------------------------------------
// Selective scan, fused with y*silu(z) gate and bf16 output.
// Block = 256 channels of one batch; B/C vectors broadcast via LDS per step.
// ---------------------------------------------------------------------------
__global__ void scan_kernel(const float* __restrict__ xdbl,  // [M,96]
                            const float* __restrict__ dt,    // [M,1024]
                            const float* __restrict__ u,     // [M,1024]
                            const float* __restrict__ xz,    // [M,2048], z at +1024
                            const float* __restrict__ Alog,  // [1024,16]
                            const float* __restrict__ Dp,    // [1024]
                            unsigned short* __restrict__ yb, // [M,1024] bf16
                            int rev) {
    const int b = blockIdx.x >> 2;
    const int d = (blockIdx.x & 3) * 256 + threadIdx.x;
    float A[16], h[16];
#pragma unroll
    for (int n = 0; n < 16; ++n) {
        A[n] = -__expf(Alog[d * 16 + n]);
        h[n] = 0.f;
    }
    const float Dd = Dp[d];
    __shared__ float sB[16], sC[16];
    for (int s = 0; s < SEQL; ++s) {
        int t = rev ? (SEQL - 1 - s) : s;
        size_t row = (size_t)b * SEQL + t;
        __syncthreads();
        if (threadIdx.x < 32) {
            float v = xdbl[row * 96 + 64 + threadIdx.x];
            if (threadIdx.x < 16) sB[threadIdx.x] = v;
            else                  sC[threadIdx.x - 16] = v;
        }
        __syncthreads();
        float dtv = dt[row * 1024 + d];
        float uv  = u[row * 1024 + d];
        float du  = dtv * uv;
        float y = 0.f;
#pragma unroll
        for (int n = 0; n < 16; ++n) {
            float dA = __expf(dtv * A[n]);
            h[n] = dA * h[n] + du * sB[n];
            y += h[n] * sC[n];
        }
        y += uv * Dd;
        float z = xz[row * 2048 + 1024 + d];
        yb[row * 1024 + d] = f32_to_bf16(y * siluf(z));
    }
}

// ---------------------------------------------------------------------------
// Host-side orchestration
// ---------------------------------------------------------------------------
extern "C" void kernel_launch(void* const* d_in, const int* in_sizes, int n_in,
                              void* d_out, int out_size, void* d_ws, size_t ws_size,
                              hipStream_t stream) {
    (void)in_sizes; (void)n_in; (void)out_size; (void)ws_size;
    const float* x      = (const float*)d_in[0];
    const float* norm_g = (const float*)d_in[1];
    const float* norm_b = (const float*)d_in[2];
    const float* in_w   = (const float*)d_in[3];
    const float* conv_w = (const float*)d_in[4];
    const float* conv_b = (const float*)d_in[5];
    const float* out_w  = (const float*)d_in[6];
    struct DirP { const float *in_w, *conv_w, *conv_b, *xp_w, *dt_w, *dt_b, *Alog, *D, *out_w; };
    DirP dirs[2];
    for (int p = 0; p < 2; ++p) {
        int o = 7 + p * 9;
        dirs[p] = { (const float*)d_in[o],     (const float*)d_in[o + 1],
                    (const float*)d_in[o + 2], (const float*)d_in[o + 3],
                    (const float*)d_in[o + 4], (const float*)d_in[o + 5],
                    (const float*)d_in[o + 6], (const float*)d_in[o + 7],
                    (const float*)d_in[o + 8] };
    }

    // ---- workspace carve (256B aligned) ----
    char* wp = (char*)d_ws;
    auto carve = [&](size_t bytes) -> void* {
        void* r = (void*)wp;
        wp += (bytes + 255) & ~(size_t)255;
        return r;
    };
    const size_t M = MROWS;
    unsigned short* w_in  = (unsigned short*)carve((size_t)2048 * 512 * 2);
    unsigned short* w_out = (unsigned short*)carve((size_t)512 * 2048 * 2);
    unsigned short *w_dinw[2], *w_dxpw[2], *w_ddtw[2], *w_doutw[2];
    for (int p = 0; p < 2; ++p) {
        w_dinw[p]  = (unsigned short*)carve((size_t)2048 * 1024 * 2);
        w_dxpw[p]  = (unsigned short*)carve((size_t)96 * 1024 * 2);
        w_ddtw[p]  = (unsigned short*)carve((size_t)1024 * 64 * 2);
        w_doutw[p] = (unsigned short*)carve((size_t)1024 * 1024 * 2);
    }
    unsigned short* XN   = (unsigned short*)carve(M * 512 * 2);
    float*          XP   = (float*)         carve(M * 2048 * 4);  // xp, then xz per dir
    float*          XA   = (float*)         carve(M * 1024 * 4);
    unsigned short* XAB  = (unsigned short*)carve(M * 1024 * 2);
    float*          G    = (float*)         carve(M * 1024 * 4);
    float*          XM   = (float*)         carve(M * 1024 * 4);
    unsigned short* XMB  = (unsigned short*)carve(M * 1024 * 2);
    float*          XDBL = (float*)         carve(M * 96 * 4);
    unsigned short* DTRB = (unsigned short*)carve(M * 64 * 2);
    float*          DT   = (float*)         carve(M * 1024 * 4);
    unsigned short* YBv  = (unsigned short*)carve(M * 1024 * 2);
    unsigned short* YM   = (unsigned short*)carve(M * 2048 * 2);

    auto cvt = [&](const float* s, unsigned short* d, int n) {
        cvt_bf16_kernel<<<(n + 255) / 256, 256, 0, stream>>>(s, d, n);
    };
    cvt(in_w,  w_in,  2048 * 512);
    cvt(out_w, w_out, 512 * 2048);
    for (int p = 0; p < 2; ++p) {
        cvt(dirs[p].in_w,  w_dinw[p],  2048 * 1024);
        cvt(dirs[p].xp_w,  w_dxpw[p],  96 * 1024);
        cvt(dirs[p].dt_w,  w_ddtw[p],  1024 * 64);
        cvt(dirs[p].out_w, w_doutw[p], 1024 * 1024);
    }

    const int MB128 = (int)(M / 128);  // 64 block rows for LDS GEMM
    const int MB64  = (int)(M / 64);   // 128 block rows for direct GEMM

    // 1) LayerNorm -> bf16
    layernorm_bf16_kernel<<<(int)M, 256, 0, stream>>>(x, norm_g, norm_b, XN);
    // 2) xp = xn @ in_w^T   (M x 2048, K=512)
    gemm_wmma_lds_kernel<0><<<dim3(2048 / 64, MB128), 256, 0, stream>>>(
        XN, w_in, XP, nullptr, nullptr, (int)M, 2048, 512, 0, 0);
    // 3) g = silu(res)
    silu_half_kernel<<<(int)(M * 1024 / 256), 256, 0, stream>>>(XP, G);
    // 4) xa = silu(dwconv(xc))
    dwconv_silu_kernel<<<(int)(M * 1024 / 256), 256, 0, stream>>>(
        XP, 2048, conv_w, conv_b, XA, XAB, 0);

    for (int p = 0; p < 2; ++p) {
        int rev = p;
        // 5) xz = xa @ dir_in_w^T  (M x 2048, K=1024) -> reuse XP
        gemm_wmma_lds_kernel<0><<<dim3(2048 / 64, MB128), 256, 0, stream>>>(
            XAB, w_dinw[p], XP, nullptr, nullptr, (int)M, 2048, 1024, 0, 0);
        // 6) xm = silu(dwconv_dir(xz[:, :1024]))
        dwconv_silu_kernel<<<(int)(M * 1024 / 256), 256, 0, stream>>>(
            XP, 2048, dirs[p].conv_w, dirs[p].conv_b, XM, XMB, rev);
        // 7) xdbl = xm @ xp_w^T  (M x 96, K=1024) -- small-N direct kernel
        gemm_wmma_kernel<2, 0><<<dim3(96 / 32, MB64), 128, 0, stream>>>(
            XMB, w_dxpw[p], XDBL, nullptr, nullptr, (int)M, 96, 1024, 0, 0);
        // 8) dtr -> bf16
        extract_dtr_kernel<<<(int)(M * 64 / 256), 256, 0, stream>>>(XDBL, DTRB);
        // 9) dt = softplus(dtr @ dt_w^T + dt_b)  (M x 1024, K=64)
        gemm_wmma_lds_kernel<3><<<dim3(1024 / 64, MB128), 256, 0, stream>>>(
            DTRB, w_ddtw[p], DT, nullptr, dirs[p].dt_b, (int)M, 1024, 64, 0, 0);
        // 10) selective scan (+ * silu(z)) -> bf16
        scan_kernel<<<16, 256, 0, stream>>>(XDBL, DT, XM, XP,
                                            dirs[p].Alog, dirs[p].D, YBv, rev);
        // 11) ym[:, p*1024:+1024] = bf16((y @ out_w^T) * g)
        gemm_wmma_lds_kernel<1><<<dim3(1024 / 64, MB128), 256, 0, stream>>>(
            YBv, w_doutw[p], nullptr, YM, G, (int)M, 1024, 1024, 2048, p * 1024);
    }
    // 12) out = ym @ out_w^T + x   (M x 512, K=2048)
    gemm_wmma_lds_kernel<2><<<dim3(512 / 64, MB128), 256, 0, stream>>>(
        YM, w_out, (float*)d_out, nullptr, x, (int)M, 512, 2048, 0, 0);
}
